// NonparametricAttentionPooling_41188736369105
// MI455X (gfx1250) — matrix operand order repaired and checked by reading, not compiled, not run on hardware
//
#include <hip/hip_runtime.h>
#include <hip/hip_bf16.h>

// Problem constants (from reference setup_inputs)
#define BB 4
#define CC 64
#define NN 4096

typedef __attribute__((ext_vector_type(16))) _Float16 v16h;
typedef __attribute__((ext_vector_type(8)))  float    v8f;

struct __align__(16) H8 { _Float16 h[8]; };

// Build a v16h fragment from two contiguous 8-half runs (16B-aligned each).
// Matches CDNA5 16-bit A/B fragment layouts: elements 0..7 = first K run,
// elements 8..15 = second K run.
__device__ __forceinline__ v16h make_frag(const _Float16* p0, const _Float16* p1) {
    H8 a = *(const H8*)p0;
    H8 b = *(const H8*)p1;
    v16h f;
#pragma unroll
    for (int i = 0; i < 8; ++i) { f[i] = a.h[i]; f[i + 8] = b.h[i]; }
    return f;
}

__device__ __forceinline__ v8f wmma_f16(v16h a, v16h b, v8f c) {
    return __builtin_amdgcn_wmma_f32_16x16x32_f16(false, a, false, b, (short)0, c, false, false);
}

// ---------------------------------------------------------------------------
// Kernel 1: transpose [B,C,N] -> [B,N,C], split fp32 into f16 hi/lo, sq norms
// ---------------------------------------------------------------------------
__global__ void prep_kernel(const float* __restrict__ x,
                            _Float16* __restrict__ xh,
                            _Float16* __restrict__ xl,
                            float* __restrict__ sq) {
    int t = blockIdx.x * blockDim.x + threadIdx.x;   // 0 .. B*N-1
    if (t >= BB * NN) return;
    int b = t / NN;
    int n = t - b * NN;
    const float* xp = x + (size_t)b * CC * NN + n;
    _Float16* oh = xh + (size_t)t * CC;
    _Float16* ol = xl + (size_t)t * CC;
    float s = 0.0f;
#pragma unroll 8
    for (int c = 0; c < CC; ++c) {
        float v = xp[(size_t)c * NN];
        s = fmaf(v, v, s);
        _Float16 hi = (_Float16)v;
        oh[c] = hi;
        ol[c] = (_Float16)(v - (float)hi);
    }
    sq[t] = s;
}

// ---------------------------------------------------------------------------
// Kernel 2: fused attention pooling (WMMA). Each wave: one 16-row tile.
// Workgroup = 4 waves (128 threads) covering 64 rows; loops over j in 32-wide
// blocks staged in LDS (row-major + transposed copies, f16 hi/lo split).
// ---------------------------------------------------------------------------
#define WAVES 4
#define BJ_STRIDE  72   // 64 halves + pad (144 B, 16B multiple)
#define BT_STRIDE  48   // 32 halves + pad (96 B, 16B multiple)
#define WS_STRIDE  48   // 32 halves + pad

__global__ void __launch_bounds__(128, 1)
attention_kernel(const _Float16* __restrict__ xth,
                 const _Float16* __restrict__ xtl,
                 const float* __restrict__ sqn,
                 float* __restrict__ nf) {
    __shared__ __align__(16) _Float16 s_bj_hi[32 * BJ_STRIDE];
    __shared__ __align__(16) _Float16 s_bj_lo[32 * BJ_STRIDE];
    __shared__ __align__(16) _Float16 s_bt_hi[64 * BT_STRIDE];
    __shared__ __align__(16) _Float16 s_bt_lo[64 * BT_STRIDE];
    __shared__ __align__(16) _Float16 s_w_hi[WAVES * 16 * WS_STRIDE];
    __shared__ __align__(16) _Float16 s_w_lo[WAVES * 16 * WS_STRIDE];

    const int tid  = threadIdx.x;
    const int lane = tid & 31;
    const int wave = tid >> 5;
    const int b    = blockIdx.y;
    const int i0   = (blockIdx.x * WAVES + wave) * 16;

    const int l15 = lane & 15;
    const int lhi = lane >> 4;          // 0 or 1 (half-wave)

    const _Float16* Xh = xth + (size_t)b * NN * CC;
    const _Float16* Xl = xtl + (size_t)b * NN * CC;
    const float*    Sq = sqn + (size_t)b * NN;

    // ---- preload A fragments for this wave's 16 rows (K = C = 64 -> 2 chunks)
    const int kb = lhi * 8;
    const _Float16* ah = Xh + (size_t)(i0 + l15) * CC;
    const _Float16* al = Xl + (size_t)(i0 + l15) * CC;
    v16h a_hi0 = make_frag(ah + kb,      ah + kb + 16);
    v16h a_hi1 = make_frag(ah + 32 + kb, ah + 32 + kb + 16);
    v16h a_lo0 = make_frag(al + kb,      al + kb + 16);
    v16h a_lo1 = make_frag(al + 32 + kb, al + 32 + kb + 16);

    float si[8];
#pragma unroll
    for (int r = 0; r < 8; ++r) si[r] = Sq[i0 + r + 8 * lhi];

    v16h ones;
#pragma unroll
    for (int i = 0; i < 16; ++i) ones[i] = (_Float16)1.0f;

    v8f acc[4];
#pragma unroll
    for (int q = 0; q < 4; ++q) acc[q] = v8f{0, 0, 0, 0, 0, 0, 0, 0};
    v8f rs = v8f{0, 0, 0, 0, 0, 0, 0, 0};

    _Float16* wsh = &s_w_hi[wave * 16 * WS_STRIDE];
    _Float16* wsl = &s_w_lo[wave * 16 * WS_STRIDE];

    for (int jb = 0; jb < NN / 32; ++jb) {
        const int j0 = jb * 32;
        __syncthreads();   // protect previous iteration's LDS readers
        // ---- cooperative stage of xt[j0:j0+32, 0:64] (row-major + transposed)
#pragma unroll
        for (int rep = 0; rep < 2; ++rep) {
            int item = tid + rep * 128;        // 0..255 -> 32 rows x 8 groups
            int jr = item >> 3;
            int g  = item & 7;
            const H8* ph = (const H8*)(Xh + (size_t)(j0 + jr) * CC) + g;
            const H8* pl = (const H8*)(Xl + (size_t)(j0 + jr) * CC) + g;
            H8 vh = *ph;
            H8 vl = *pl;
            *(H8*)&s_bj_hi[jr * BJ_STRIDE + g * 8] = vh;
            *(H8*)&s_bj_lo[jr * BJ_STRIDE + g * 8] = vl;
#pragma unroll
            for (int i = 0; i < 8; ++i) {
                s_bt_hi[(g * 8 + i) * BT_STRIDE + jr] = vh.h[i];
                s_bt_lo[(g * 8 + i) * BT_STRIDE + jr] = vl.h[i];
            }
        }
        __syncthreads();

        // ---- GEMM1: Gram tile (16 x 32) -> weights, staged as f16 hi/lo
#pragma unroll
        for (int jj = 0; jj < 32; jj += 16) {
            const _Float16* bh = &s_bj_hi[(jj + l15) * BJ_STRIDE];
            const _Float16* bl = &s_bj_lo[(jj + l15) * BJ_STRIDE];
            const int ob = lhi * 8;
            float sqj = Sq[j0 + jj + l15];   // overlap with WMMAs below

            v8f c = v8f{0, 0, 0, 0, 0, 0, 0, 0};
            {   // K chunk 0 (K = 0..31)
                v16h bh0 = make_frag(bh + ob, bh + ob + 16);
                v16h bl0 = make_frag(bl + ob, bl + ob + 16);
                c = wmma_f16(a_hi0, bh0, c);
                c = wmma_f16(a_hi0, bl0, c);
                c = wmma_f16(a_lo0, bh0, c);
            }
            {   // K chunk 1 (K = 32..63)
                v16h bh1 = make_frag(bh + 32 + ob, bh + 32 + ob + 16);
                v16h bl1 = make_frag(bl + 32 + ob, bl + 32 + ob + 16);
                c = wmma_f16(a_hi1, bh1, c);
                c = wmma_f16(a_hi1, bl1, c);
                c = wmma_f16(a_lo1, bh1, c);
            }

#pragma unroll
            for (int r = 0; r < 8; ++r) {
                float d2 = si[r] + sqj - 2.0f * c[r];
                d2 = fmaxf(d2, 0.0f);
                // bandwidth = 1:  0.5*exp(-d2/2) + 0.3*exp(-d2/8) + 0.2*exp(-2 d2)
                float w = 0.5f * __expf(-0.5f * d2)
                        + 0.3f * __expf(-0.125f * d2)
                        + 0.2f * __expf(-2.0f * d2);
                _Float16 whf = (_Float16)w;
                _Float16 wlf = (_Float16)(w - (float)whf);
                int m = r + 8 * lhi;
                wsh[m * WS_STRIDE + jj + l15] = whf;
                wsl[m * WS_STRIDE + jj + l15] = wlf;
            }
        }

        // ---- GEMM2: acc += W @ xt_j  (plus ones-column WMMA for row sums)
        {
            const _Float16* pwh = &wsh[l15 * WS_STRIDE];
            const _Float16* pwl = &wsl[l15 * WS_STRIDE];
            v16h w_hi = make_frag(pwh + kb, pwh + kb + 16);
            v16h w_lo = make_frag(pwl + kb, pwl + kb + 16);

            rs = wmma_f16(w_hi, ones, rs);
            rs = wmma_f16(w_lo, ones, rs);

#pragma unroll
            for (int q = 0; q < 4; ++q) {
                const int crow = q * 16 + l15;
                const int ko = lhi * 8;
                v16h b2h = make_frag(&s_bt_hi[crow * BT_STRIDE + ko],
                                     &s_bt_hi[crow * BT_STRIDE + ko + 16]);
                v16h b2l = make_frag(&s_bt_lo[crow * BT_STRIDE + ko],
                                     &s_bt_lo[crow * BT_STRIDE + ko + 16]);
                acc[q] = wmma_f16(w_hi, b2h, acc[q]);
                acc[q] = wmma_f16(w_hi, b2l, acc[q]);
                acc[q] = wmma_f16(w_lo, b2h, acc[q]);
            }
        }
    }

    // ---- normalize rows and write nf [B,N,C]
#pragma unroll
    for (int r = 0; r < 8; ++r) {
        int m = r + 8 * lhi;
        float inv = 1.0f / (rs[r] + 1e-8f);
        size_t row = (size_t)(b * NN + i0 + m) * CC;
#pragma unroll
        for (int q = 0; q < 4; ++q) {
            nf[row + q * 16 + l15] = acc[q][r] * inv;
        }
    }
}

// ---------------------------------------------------------------------------
// Kernel 3: per-channel BN statistics over (B, N)
// ---------------------------------------------------------------------------
__global__ void bnstats_kernel(const float* __restrict__ nf,
                               float* __restrict__ stats) {
    __shared__ float sh1[256];
    __shared__ float sh2[256];
    const int c = blockIdx.x;
    float s = 0.0f, s2 = 0.0f;
    for (int i = threadIdx.x; i < BB * NN; i += 256) {
        float v = nf[(size_t)i * CC + c];
        s += v;
        s2 = fmaf(v, v, s2);
    }
    sh1[threadIdx.x] = s;
    sh2[threadIdx.x] = s2;
    __syncthreads();
    for (int off = 128; off > 0; off >>= 1) {
        if (threadIdx.x < off) {
            sh1[threadIdx.x] += sh1[threadIdx.x + off];
            sh2[threadIdx.x] += sh2[threadIdx.x + off];
        }
        __syncthreads();
    }
    if (threadIdx.x == 0) {
        const float invn = 1.0f / (float)(BB * NN);
        float mean = sh1[0] * invn;
        float var  = fmaxf(sh2[0] * invn - mean * mean, 0.0f);
        stats[c]       = mean;
        stats[CC + c]  = rsqrtf(var + 1e-5f);
    }
}

// ---------------------------------------------------------------------------
// Kernel 4: normalize + exact GELU, write [B,C,N]
// ---------------------------------------------------------------------------
__global__ void finalize_kernel(const float* __restrict__ nf,
                                const float* __restrict__ stats,
                                float* __restrict__ out) {
    int idx = blockIdx.x * blockDim.x + threadIdx.x;
    if (idx >= BB * CC * NN) return;
    int n = idx & (NN - 1);
    int c = (idx / NN) & (CC - 1);
    int b = idx / (NN * CC);
    float v = nf[(size_t)(b * NN + n) * CC + c];
    float y = (v - stats[c]) * stats[CC + c];
    out[idx] = 0.5f * y * (1.0f + erff(y * 0.70710678118654752f));
}

// ---------------------------------------------------------------------------
extern "C" void kernel_launch(void* const* d_in, const int* in_sizes, int n_in,
                              void* d_out, int out_size, void* d_ws, size_t ws_size,
                              hipStream_t stream) {
    (void)in_sizes; (void)n_in; (void)out_size; (void)ws_size;
    const float* x = (const float*)d_in[0];
    float* out = (float*)d_out;

    // workspace layout
    char* w = (char*)d_ws;
    _Float16* xt_hi = (_Float16*)(w);                              // 2 MB
    _Float16* xt_lo = (_Float16*)(w + (size_t)2 * 1024 * 1024);    // 2 MB
    float*    sq    = (float*)   (w + (size_t)4 * 1024 * 1024);    // 64 KB
    float*    nf    = (float*)   (w + (size_t)4 * 1024 * 1024 + 65536);  // 4 MB
    float*    stats = (float*)   (w + (size_t)8 * 1024 * 1024 + 65536);  // 512 B

    prep_kernel<<<(BB * NN + 255) / 256, 256, 0, stream>>>(x, xt_hi, xt_lo, sq);

    attention_kernel<<<dim3(NN / (16 * WAVES), BB), 128, 0, stream>>>(
        xt_hi, xt_lo, sq, nf);

    bnstats_kernel<<<CC, 256, 0, stream>>>(nf, stats);

    finalize_kernel<<<(BB * CC * NN + 255) / 256, 256, 0, stream>>>(nf, stats, out);
}
